// Block_9672266350877
// MI455X (gfx1250) — compile-verified, matrix operand order checked by reading, not run on hardware
//
#include <hip/hip_runtime.h>
#include <hip/hip_bf16.h>
#include <math.h>
#include <type_traits>

// ---------------------------------------------------------------------------
// Types for CDNA5 WMMA
// ---------------------------------------------------------------------------
typedef __bf16 bf16_t;
typedef __attribute__((ext_vector_type(16))) __bf16 v16bf;
typedef __attribute__((ext_vector_type(8)))  __bf16 v8bf;
typedef __attribute__((ext_vector_type(8)))  float  v8f;

union BV {
    v16bf v;
    v8bf  h[2];
};

__device__ __forceinline__ v8f wmma_bf16(v16bf a, v16bf b, v8f c) {
    // D = A(16x32 bf16) * B(32x16 bf16) + C(16x16 f32)
    return __builtin_amdgcn_wmma_f32_16x16x32_bf16(
        /*neg_a=*/false, a, /*neg_b=*/false, b,
        /*c_mod=*/(short)0, c, /*reuse_a=*/false, /*reuse_b=*/false);
}

// ---------------------------------------------------------------------------
// CDNA5 async global -> LDS copy (ASYNCcnt-tracked DMA, bypasses VGPRs).
// Generic pointers to LDS carry the LDS byte offset in addr[31:0] (ISA 10.2).
// ---------------------------------------------------------------------------
__device__ __forceinline__ void async_copy_b128(void* lds_dst, const void* gsrc) {
    unsigned int lds_off = (unsigned int)(unsigned long long)lds_dst;
    asm volatile("global_load_async_to_lds_b128 %0, %1, off"
                 :: "v"(lds_off), "v"(gsrc) : "memory");
}

__device__ __forceinline__ void wait_async0() {
#if __has_builtin(__builtin_amdgcn_s_wait_asynccnt)
    __builtin_amdgcn_s_wait_asynccnt(0);
#else
    asm volatile("s_wait_asynccnt 0x0" ::: "memory");
#endif
}

// ---------------------------------------------------------------------------
// 16-lane (DPP "row") butterfly reductions, pure VALU, no LDS round trips.
// Groups are lanes 0-15 / 16-31, which DPP row ops never cross.
// ---------------------------------------------------------------------------
#if __has_builtin(__builtin_amdgcn_update_dpp)
template <int CTRL>
__device__ __forceinline__ float dpp_mov(float x) {
    return __int_as_float(
        __builtin_amdgcn_update_dpp(0, __float_as_int(x), CTRL, 0xF, 0xF, true));
}
__device__ __forceinline__ float red_max16(float x) {
    x = fmaxf(x, dpp_mov<0xB1>(x));   // quad_perm [1,0,3,2]  : xor 1
    x = fmaxf(x, dpp_mov<0x4E>(x));   // quad_perm [2,3,0,1]  : xor 2
    x = fmaxf(x, dpp_mov<0x141>(x));  // row_half_mirror      : xor 4
    x = fmaxf(x, dpp_mov<0x140>(x));  // row_mirror           : xor 8
    return x;
}
__device__ __forceinline__ float red_sum16(float x) {
    x += dpp_mov<0xB1>(x);
    x += dpp_mov<0x4E>(x);
    x += dpp_mov<0x141>(x);
    x += dpp_mov<0x140>(x);
    return x;
}
#else
__device__ __forceinline__ float red_max16(float x) {
    #pragma unroll
    for (int off = 8; off; off >>= 1) x = fmaxf(x, __shfl_xor(x, off, 32));
    return x;
}
__device__ __forceinline__ float red_sum16(float x) {
    #pragma unroll
    for (int off = 8; off; off >>= 1) x += __shfl_xor(x, off, 32);
    return x;
}
#endif

static constexpr int Bsz = 2;
static constexpr int T   = 2048;
static constexpr int C   = 1024;
static constexpr int H   = 16;
static constexpr int HD  = 64;
static constexpr int M   = Bsz * T;   // 4096 rows

// ---------------------------------------------------------------------------
// fp32 -> bf16 conversion (weights)
// ---------------------------------------------------------------------------
__global__ void cvt_f32_bf16(const float* __restrict__ src,
                             bf16_t* __restrict__ dst, int n) {
    int i = blockIdx.x * blockDim.x + threadIdx.x;
    if (i < n) dst[i] = (bf16_t)src[i];
}

// ---------------------------------------------------------------------------
// LayerNorm over C=1024, one block (256 threads) per row, bf16 output
// ---------------------------------------------------------------------------
__global__ __launch_bounds__(256)
void ln_kernel(const float* __restrict__ x, const float* __restrict__ w,
               const float* __restrict__ b, bf16_t* __restrict__ out) {
    int row  = blockIdx.x;
    int tid  = threadIdx.x;
    int wave = tid >> 5, lane = tid & 31;

    const float* xr = x + (size_t)row * C;
    float4 xv = ((const float4*)xr)[tid];
    float s1 = xv.x + xv.y + xv.z + xv.w;
    float s2 = xv.x*xv.x + xv.y*xv.y + xv.z*xv.z + xv.w*xv.w;
    #pragma unroll
    for (int off = 16; off; off >>= 1) {
        s1 += __shfl_xor(s1, off, 32);
        s2 += __shfl_xor(s2, off, 32);
    }
    __shared__ float a1[8], a2[8];
    if (lane == 0) { a1[wave] = s1; a2[wave] = s2; }
    __syncthreads();
    float t1 = 0.f, t2 = 0.f;
    #pragma unroll
    for (int i = 0; i < 8; ++i) { t1 += a1[i]; t2 += a2[i]; }
    float mu   = t1 * (1.0f / C);
    float var  = t2 * (1.0f / C) - mu * mu;
    float rstd = rsqrtf(var + 1e-5f);

    float4 wv = ((const float4*)w)[tid];
    float4 bv = ((const float4*)b)[tid];
    bf16_t* orow = out + (size_t)row * C;
    orow[tid*4 + 0] = (bf16_t)((xv.x - mu) * rstd * wv.x + bv.x);
    orow[tid*4 + 1] = (bf16_t)((xv.y - mu) * rstd * wv.y + bv.y);
    orow[tid*4 + 2] = (bf16_t)((xv.z - mu) * rstd * wv.z + bv.z);
    orow[tid*4 + 3] = (bf16_t)((xv.w - mu) * rstd * wv.w + bv.w);
}

// ---------------------------------------------------------------------------
// GEMM: out[m,n] = epi( sum_k A[m,k] * W[n,k] + bias[n] )   (A @ W^T)
// Block = 256 threads (8 waves); block tile 256(M) x 64(N); K-step 32.
// Each wave owns a 32x64 strip (2 M-tiles x 4 N-tiles = 8 WMMAs / k-step).
// The shared 64x32 W tile is staged into LDS with double-buffered async
// global->LDS DMA (ASYNCcnt) and consumed via ds_load_b128 in the CDNA5
// 16-bit operand interleave:
//   lanes 0-15  hold K [0..7] and [16..23]
//   lanes 16-31 hold K [8..15] and [24..31]
// EPI: 0 = bias -> bf16 ; 1 = bias + residual -> f32 ; 2 = bias + GELU -> bf16
// ---------------------------------------------------------------------------
template <int EPI>
__global__ __launch_bounds__(256)
void gemm_bf16(const bf16_t* __restrict__ A, const bf16_t* __restrict__ W,
               const float* __restrict__ bias, const float* __restrict__ resid,
               void* __restrict__ out, int N, int K) {
    __shared__ __align__(16) bf16_t wt[2][64 * 32];   // double-buffered W tile

    int tid  = threadIdx.x;
    int wave = tid >> 5;
    int lane = tid & 31;
    int r    = lane & 15;
    int sel  = lane >> 4;        // which 16-lane half

    int m0 = blockIdx.x * 256 + wave * 32;
    int n0 = blockIdx.y * 64;

    // cooperative async W-tile fetch: 256 threads x 16B = 64 rows x 32 bf16
    int wrow  = tid >> 2;        // 0..63
    int wchk  = tid & 3;         // 0..3, 8 bf16 each
    const bf16_t* wsrc0 = W + (size_t)(n0 + wrow) * K + wchk * 8;
    bf16_t*       wdst0 = &wt[0][wrow * 32 + wchk * 8];
    bf16_t*       wdst1 = &wt[1][wrow * 32 + wchk * 8];

    async_copy_b128(wdst0, wsrc0);                    // prologue: k-step 0

    v8f acc[2][4] = {};
    const bf16_t* arow0 = A + (size_t)(m0 + r) * K;
    const bf16_t* arow1 = A + (size_t)(m0 + 16 + r) * K;

    for (int kk = 0; kk < K; kk += 32) {
        wait_async0();           // current buffer resident (this wave's issues)
        __syncthreads();         // ... and everyone else's

        int cur = (kk >> 5) & 1;
        if (kk + 32 < K)         // overlap next tile's DMA with this step's math
            async_copy_b128(cur ? wdst0 : wdst1, wsrc0 + kk + 32);

        if (kk + 256 < K) {
            __builtin_prefetch(arow0 + kk + 256, 0, 0);
            __builtin_prefetch(arow1 + kk + 256, 0, 0);
        }
        BV a0, a1;
        a0.h[0] = *(const v8bf*)(arow0 + kk + sel * 8);
        a0.h[1] = *(const v8bf*)(arow0 + kk + 16 + sel * 8);
        a1.h[0] = *(const v8bf*)(arow1 + kk + sel * 8);
        a1.h[1] = *(const v8bf*)(arow1 + kk + 16 + sel * 8);

        const bf16_t* wbuf = wt[cur];
        #pragma unroll
        for (int nt = 0; nt < 4; ++nt) {
            const bf16_t* brow = wbuf + (nt * 16 + r) * 32;
            BV bb;
            bb.h[0] = *(const v8bf*)(brow + sel * 8);
            bb.h[1] = *(const v8bf*)(brow + 16 + sel * 8);
            acc[0][nt] = wmma_bf16(a0.v, bb.v, acc[0][nt]);
            acc[1][nt] = wmma_bf16(a1.v, bb.v, acc[1][nt]);
        }
        // no trailing barrier: this buffer's ds_loads are drained before the
        // wmma above, and its next overwrite is issued only after the next
        // top-of-loop barrier.
    }

    // C/D layout: VGPR j -> row +j + 8*sel ; column n0 + nt*16 + (lane&15)
    #pragma unroll
    for (int mt = 0; mt < 2; ++mt) {
        #pragma unroll
        for (int nt = 0; nt < 4; ++nt) {
            int col = n0 + nt * 16 + r;
            float bcol = bias[col];
            #pragma unroll
            for (int j = 0; j < 8; ++j) {
                int row = m0 + mt * 16 + j + sel * 8;
                size_t idx = (size_t)row * N + col;
                float v = acc[mt][nt][j] + bcol;
                if (EPI == 0) {
                    ((bf16_t*)out)[idx] = (bf16_t)v;
                } else if (EPI == 1) {
                    ((float*)out)[idx] = resid[idx] + v;
                } else {
                    float g = 0.5f * v * (1.0f + erff(v * 0.70710678118f));
                    ((bf16_t*)out)[idx] = (bf16_t)g;
                }
            }
        }
    }
}

// ---------------------------------------------------------------------------
// Causal flash attention. q,k,v,y: (B,T,C) bf16 with heads packed (h*HD+d).
// Block = 128 threads (4 waves); block owns 64 query rows of one (b,h);
// each wave owns 16 query rows, iterates keys 32 at a time. V tiles are
// pulled into wave-private LDS with async global->LDS DMA; softmax row
// statistics use DPP butterfly reductions; q is pre-scaled by 1/sqrt(HD)
// (exact in bf16: power of two); causal masking only runs on the one tail
// block that straddles the diagonal.
// ---------------------------------------------------------------------------
__global__ __launch_bounds__(128)
void attn_kernel(const bf16_t* __restrict__ q, const bf16_t* __restrict__ k,
                 const bf16_t* __restrict__ v, bf16_t* __restrict__ y) {
    __shared__ __align__(16) bf16_t pbuf[4][16 * 32];  // P tile per wave
    __shared__ __align__(16) bf16_t vbuf[4][32 * 64];  // V tile per wave (k-major)

    int wave = threadIdx.x >> 5;
    int lane = threadIdx.x & 31;
    int r    = lane & 15;
    int sel  = lane >> 4;

    int bq = blockIdx.x & 31;          // T/64 = 32 query blocks
    int hh = (blockIdx.x >> 5) & 15;   // head
    int bb = blockIdx.x >> 9;          // batch
    int qbase = bq * 64 + wave * 16;

    size_t base = (size_t)bb * T * C + (size_t)hh * HD;

    // q A-operands for the two K=32 chunks of HD=64, pre-scaled by 1/8
    BV qa[2];
    const bf16_t* qrow = q + base + (size_t)(qbase + r) * C;
    #pragma unroll
    for (int c = 0; c < 2; ++c) {
        qa[c].h[0] = *(const v8bf*)(qrow + c * 32 + sel * 8);
        qa[c].h[1] = *(const v8bf*)(qrow + c * 32 + 16 + sel * 8);
        #pragma unroll
        for (int i = 0; i < 16; ++i)
            qa[c].v[i] = (bf16_t)((float)qa[c].v[i] * 0.125f);
    }

    v8f o[4] = {};
    float mrow[8], lrow[8];
    #pragma unroll
    for (int j = 0; j < 8; ++j) { mrow[j] = -INFINITY; lrow[j] = 0.f; }

    bf16_t* ps = pbuf[wave];
    bf16_t* vs = vbuf[wave];

    auto process_block = [&](int kb, auto maskc) {
        constexpr bool MASK = decltype(maskc)::value;

        // ---- async-stage V tile (32 keys x 64 dims), k-major, wave-private.
        // Prior reads of vs are already drained (wmma consumed them), so the
        // DMA writes issued now cannot race them.
        {
            const bf16_t* vr = v + base + (size_t)(kb + lane) * C;
            #pragma unroll
            for (int c = 0; c < 8; ++c)
                async_copy_b128(vs + lane * 64 + c * 8, vr + c * 8);
        }

        // ---- S = q @ k^T for 32 keys (two 16x16 tiles) ----
        v8f s0 = {}, s1 = {};
        #pragma unroll
        for (int c = 0; c < 2; ++c) {
            const bf16_t* k0 = k + base + (size_t)(kb + r) * C + c * 32;
            BV kop;
            kop.h[0] = *(const v8bf*)(k0 + sel * 8);
            kop.h[1] = *(const v8bf*)(k0 + 16 + sel * 8);
            s0 = wmma_bf16(qa[c].v, kop.v, s0);
            const bf16_t* k1 = k + base + (size_t)(kb + 16 + r) * C + c * 32;
            kop.h[0] = *(const v8bf*)(k1 + sel * 8);
            kop.h[1] = *(const v8bf*)(k1 + 16 + sel * 8);
            s1 = wmma_bf16(qa[c].v, kop.v, s1);
        }

        // ---- online softmax per row (DPP 16-lane butterflies) ----
        #pragma unroll
        for (int j = 0; j < 8; ++j) {
            float a0 = s0[j], a1 = s1[j];
            if (MASK) {
                int qi = qbase + j + sel * 8;
                a0 = (kb + r      <= qi) ? a0 : -INFINITY;
                a1 = (kb + 16 + r <= qi) ? a1 : -INFINITY;
            }
            float mt = red_max16(fmaxf(a0, a1));
            float mnew = fmaxf(mrow[j], mt);
            float p0 = __expf(a0 - mnew);
            float p1 = __expf(a1 - mnew);
            float rs = red_sum16(p0 + p1);
            float alpha = __expf(mrow[j] - mnew);
            lrow[j] = lrow[j] * alpha + rs;
            mrow[j] = mnew;
            #pragma unroll
            for (int nt = 0; nt < 4; ++nt) o[nt][j] *= alpha;
            int prow = j + sel * 8;
            ps[prow * 32 + r]      = (bf16_t)p0;
            ps[prow * 32 + 16 + r] = (bf16_t)p1;
        }

        // ---- O += P @ V (one 16x16x32 wmma per 16-wide d tile) ----
        BV pa;
        pa.h[0] = *(const v8bf*)(ps + r * 32 + sel * 8);
        pa.h[1] = *(const v8bf*)(ps + r * 32 + 16 + sel * 8);
        wait_async0();           // V tile resident (wave-private: no barrier)
        #pragma unroll
        for (int nt = 0; nt < 4; ++nt) {
            BV vb;
            #pragma unroll
            for (int i = 0; i < 8; ++i) {
                vb.v[i]     = vs[(sel * 8 + i) * 64 + nt * 16 + r];
                vb.v[8 + i] = vs[(16 + sel * 8 + i) * 64 + nt * 16 + r];
            }
            o[nt] = wmma_bf16(pa.v, vb.v, o[nt]);
        }
    };

    int nblk  = (qbase + 16 + 31) / 32;  // total 32-key blocks
    int nfull = qbase >> 5;              // blocks entirely below the diagonal
    for (int jb = 0; jb < nfull; ++jb)
        process_block(jb * 32, std::integral_constant<bool, false>{});
    for (int jb = nfull; jb < nblk; ++jb)
        process_block(jb * 32, std::integral_constant<bool, true>{});

    // ---- finalize: y = O / l ----
    #pragma unroll
    for (int j = 0; j < 8; ++j) {
        int qi = qbase + j + sel * 8;
        float inv = 1.0f / lrow[j];
        #pragma unroll
        for (int nt = 0; nt < 4; ++nt)
            y[base + (size_t)qi * C + nt * 16 + r] = (bf16_t)(o[nt][j] * inv);
    }
}

// ---------------------------------------------------------------------------
// Host-side launch
// ---------------------------------------------------------------------------
extern "C" void kernel_launch(void* const* d_in, const int* in_sizes, int n_in,
                              void* d_out, int out_size, void* d_ws, size_t ws_size,
                              hipStream_t stream) {
    const float* x     = (const float*)d_in[0];
    const float* ln1_w = (const float*)d_in[1];
    const float* ln1_b = (const float*)d_in[2];
    const float* ln2_w = (const float*)d_in[3];
    const float* ln2_b = (const float*)d_in[4];
    const float* Wq = (const float*)d_in[5];  const float* bq = (const float*)d_in[6];
    const float* Wk = (const float*)d_in[7];  const float* bk = (const float*)d_in[8];
    const float* Wv = (const float*)d_in[9];  const float* bv = (const float*)d_in[10];
    const float* Wp = (const float*)d_in[11]; const float* bp = (const float*)d_in[12];
    const float* W1 = (const float*)d_in[13]; const float* b1 = (const float*)d_in[14];
    const float* W2 = (const float*)d_in[15]; const float* b2 = (const float*)d_in[16];
    float* outp = (float*)d_out;

    // workspace layout
    char* w = (char*)d_ws;
    auto take = [&](size_t bytes) { char* p = w; w += (bytes + 255) & ~size_t(255); return p; };
    bf16_t* hb   = (bf16_t*)take((size_t)M * C * 2);
    bf16_t* wqb  = (bf16_t*)take((size_t)C * C * 2);
    bf16_t* wkb  = (bf16_t*)take((size_t)C * C * 2);
    bf16_t* wvb  = (bf16_t*)take((size_t)C * C * 2);
    bf16_t* wpb  = (bf16_t*)take((size_t)C * C * 2);
    bf16_t* w1b  = (bf16_t*)take((size_t)4 * C * C * 2);
    bf16_t* w2b  = (bf16_t*)take((size_t)4 * C * C * 2);
    bf16_t* qb   = (bf16_t*)take((size_t)M * C * 2);
    bf16_t* kbuf = (bf16_t*)take((size_t)M * C * 2);
    bf16_t* vb_  = (bf16_t*)take((size_t)M * C * 2);
    bf16_t* yb   = (bf16_t*)take((size_t)M * C * 2);
    float*  xres = (float*) take((size_t)M * C * 4);
    bf16_t* h2b  = (bf16_t*)take((size_t)M * C * 2);
    bf16_t* mact = (bf16_t*)take((size_t)M * 4 * C * 2);

    // 1) weights -> bf16
    int nw = C * C;
    cvt_f32_bf16<<<(nw + 255) / 256, 256, 0, stream>>>(Wq, wqb, nw);
    cvt_f32_bf16<<<(nw + 255) / 256, 256, 0, stream>>>(Wk, wkb, nw);
    cvt_f32_bf16<<<(nw + 255) / 256, 256, 0, stream>>>(Wv, wvb, nw);
    cvt_f32_bf16<<<(nw + 255) / 256, 256, 0, stream>>>(Wp, wpb, nw);
    cvt_f32_bf16<<<(4 * nw + 255) / 256, 256, 0, stream>>>(W1, w1b, 4 * nw);
    cvt_f32_bf16<<<(4 * nw + 255) / 256, 256, 0, stream>>>(W2, w2b, 4 * nw);

    // 2) LN1
    ln_kernel<<<M, 256, 0, stream>>>(x, ln1_w, ln1_b, hb);

    // 3) QKV projections
    dim3 g1(M / 256, C / 64);
    gemm_bf16<0><<<g1, 256, 0, stream>>>(hb, wqb, bq, nullptr, qb,   C, C);
    gemm_bf16<0><<<g1, 256, 0, stream>>>(hb, wkb, bk, nullptr, kbuf, C, C);
    gemm_bf16<0><<<g1, 256, 0, stream>>>(hb, wvb, bv, nullptr, vb_,  C, C);

    // 4) causal attention
    attn_kernel<<<Bsz * H * (T / 64), 128, 0, stream>>>(qb, kbuf, vb_, yb);

    // 5) output projection + residual (f32)
    gemm_bf16<1><<<g1, 256, 0, stream>>>(yb, wpb, bp, x, xres, C, C);

    // 6) LN2
    ln_kernel<<<M, 256, 0, stream>>>(xres, ln2_w, ln2_b, h2b);

    // 7) MLP fc1 + GELU
    dim3 g2(M / 256, (4 * C) / 64);
    gemm_bf16<2><<<g2, 256, 0, stream>>>(h2b, w1b, b1, nullptr, mact, 4 * C, C);

    // 8) MLP fc2 + residual -> final output (f32)
    gemm_bf16<1><<<g1, 256, 0, stream>>>(mact, w2b, b2, xres, outp, C, 4 * C);
}
